// KNNClassifier_30288109371842
// MI455X (gfx1250) — matrix operand order repaired and checked by reading, not compiled, not run on hardware
//
#include <hip/hip_runtime.h>
#include <hip/hip_bf16.h>

// Problem constants (match reference)
#define BQ 4096
#define NTRAIN 65536
#define DIM 512
#define NUM_CLASSES 100
#define KNN_K 200
#define CHUNK 2048           // train columns per GEMM/select iteration
#define NCHUNK (NTRAIN / CHUNK)
#define TOPW 256             // running top-list width (>= KNN_K)
#define SORT_P 4096          // bitonic capacity: CHUNK + TOPW padded to pow2

typedef __bf16 v16bf __attribute__((ext_vector_type(16)));
typedef float  v8f   __attribute__((ext_vector_type(8)));

union Frag {
    v16bf v;
    uint4 q[2];
};

// ---------------------------------------------------------------------------
// fp32 -> bf16 (RNE) conversion into workspace
// ---------------------------------------------------------------------------
__global__ __launch_bounds__(256) void knn_convert_bf16(
    const float* __restrict__ src, unsigned short* __restrict__ dst, int n) {
    int i = blockIdx.x * blockDim.x + threadIdx.x;
    if (i < n) {
        unsigned u = __float_as_uint(src[i]);
        unsigned r = u + 0x7fffu + ((u >> 16) & 1u);   // round to nearest even
        dst[i] = (unsigned short)(r >> 16);
    }
}

// ---------------------------------------------------------------------------
// bf16 WMMA GEMM: sim[b, j] = sum_d x[b,d] * x_train[col0+j, d]
// WG = 256 threads (8 waves). Tile = 128(M) x 128(N). Wave = 64x32 (4x2 frags)
// ---------------------------------------------------------------------------
__global__ __launch_bounds__(256) void knn_gemm_bf16(
    const unsigned short* __restrict__ xb,   // [BQ, DIM] bf16 bits
    const unsigned short* __restrict__ tb,   // [NTRAIN, DIM] bf16 bits
    float* __restrict__ sim,                 // [BQ, CHUNK]
    int col0) {
    const int lane  = threadIdx.x & 31;
    const int wave  = threadIdx.x >> 5;
    const int waveM = wave & 1;   // 2 waves along M
    const int waveN = wave >> 1;  // 4 waves along N
    const int tileM  = blockIdx.y * 128;
    const int tileNl = blockIdx.x * 128;   // column inside chunk
    const int h  = lane >> 4;     // half-wave select
    const int mr = lane & 15;     // row (A) / col (B) within 16

    v8f acc[4][2];
#pragma unroll
    for (int mi = 0; mi < 4; ++mi)
#pragma unroll
        for (int ni = 0; ni < 2; ++ni)
            acc[mi][ni] = (v8f){0.f, 0.f, 0.f, 0.f, 0.f, 0.f, 0.f, 0.f};

    const unsigned short* arow[4];
#pragma unroll
    for (int mi = 0; mi < 4; ++mi)
        arow[mi] = xb + (size_t)(tileM + waveM * 64 + mi * 16 + mr) * DIM;
    const unsigned short* brow[2];
#pragma unroll
    for (int ni = 0; ni < 2; ++ni)
        brow[ni] = tb + (size_t)(col0 + tileNl + waveN * 32 + ni * 16 + mr) * DIM;

    for (int kk = 0; kk < DIM; kk += 32) {
        Frag a[4], b[2];
        // A 16x32 bf16 layout: lane half h -> K base kk + h*8 and kk + 16 + h*8
#pragma unroll
        for (int mi = 0; mi < 4; ++mi) {
            a[mi].q[0] = *(const uint4*)(arow[mi] + kk + h * 8);
            a[mi].q[1] = *(const uint4*)(arow[mi] + kk + 16 + h * 8);
        }
        // B 32x16 bf16 layout: lane half h -> K base kk + h*8 and kk + 16 + h*8
#pragma unroll
        for (int ni = 0; ni < 2; ++ni) {
            b[ni].q[0] = *(const uint4*)(brow[ni] + kk + h * 8);
            b[ni].q[1] = *(const uint4*)(brow[ni] + kk + 16 + h * 8);
        }
#pragma unroll
        for (int mi = 0; mi < 4; ++mi)
#pragma unroll
            for (int ni = 0; ni < 2; ++ni)
                acc[mi][ni] = __builtin_amdgcn_wmma_f32_16x16x32_bf16(
                    false, a[mi].v, false, b[ni].v,
                    (short)0, acc[mi][ni], false, false);
    }

    // C/D layout: VGPR r -> M = r + 8*h, lane%16 -> N
#pragma unroll
    for (int mi = 0; mi < 4; ++mi)
#pragma unroll
        for (int ni = 0; ni < 2; ++ni) {
            int col = tileNl + waveN * 32 + ni * 16 + mr;
#pragma unroll
            for (int r = 0; r < 8; ++r) {
                int row = tileM + waveM * 64 + mi * 16 + r + 8 * h;
                sim[(size_t)row * CHUNK + col] = acc[mi][ni][r];
            }
        }
}

// ---------------------------------------------------------------------------
// Per-row merge of a 2048-wide sim chunk with the running top-256 list via
// in-LDS bitonic sort (descending). One WG per query row.
// ---------------------------------------------------------------------------
__global__ __launch_bounds__(256) void knn_merge_topk(
    const float* __restrict__ sim,
    float* __restrict__ topv, int* __restrict__ topi,
    int col0, int first) {
    __shared__ float sv[SORT_P];
    __shared__ int   si[SORT_P];
    const int b = blockIdx.x;
    const float NEG = -3.402823466e38f;

    for (int i = threadIdx.x; i < SORT_P; i += 256) {
        float v; int ix;
        if (i < CHUNK) {
            v = sim[(size_t)b * CHUNK + i];
            ix = col0 + i;
        } else if (i < CHUNK + TOPW) {
            int j = i - CHUNK;
            if (first) { v = NEG; ix = -1; }
            else { v = topv[(size_t)b * TOPW + j]; ix = topi[(size_t)b * TOPW + j]; }
        } else {
            v = NEG; ix = -1;
        }
        sv[i] = v; si[i] = ix;
    }
    __syncthreads();

    for (int k = 2; k <= SORT_P; k <<= 1) {
        for (int j = k >> 1; j > 0; j >>= 1) {
            for (int i = threadIdx.x; i < SORT_P; i += 256) {
                int ixj = i ^ j;
                if (ixj > i) {
                    bool descBlock = ((i & k) == 0);
                    float a = sv[i], c = sv[ixj];
                    bool sw = descBlock ? (a < c) : (a > c);
                    if (sw) {
                        sv[i] = c; sv[ixj] = a;
                        int t = si[i]; si[i] = si[ixj]; si[ixj] = t;
                    }
                }
            }
            __syncthreads();
        }
    }

    topv[(size_t)b * TOPW + threadIdx.x] = sv[threadIdx.x];
    topi[(size_t)b * TOPW + threadIdx.x] = si[threadIdx.x];
}

// ---------------------------------------------------------------------------
// Weighted vote + stable descending argsort of 100 class scores.
// Deterministic (no float atomics) so graph replays are bitwise stable.
// ---------------------------------------------------------------------------
__global__ __launch_bounds__(256) void knn_finalize(
    const float* __restrict__ topv, const int* __restrict__ topi,
    const int* __restrict__ ytrain, float* __restrict__ out) {
    __shared__ float wv[KNN_K];
    __shared__ int   wl[KNN_K];
    __shared__ float cls[NUM_CLASSES];
    const int b = blockIdx.x;
    const int t = threadIdx.x;

    if (t < KNN_K) {
        float v = topv[(size_t)b * TOPW + t];
        int ix  = topi[(size_t)b * TOPW + t];
        wv[t] = __expf(v * 10.0f);            // exp(v / 0.1)
        int l = (ix >= 0) ? ytrain[ix] : 0;
        wl[t] = ((unsigned)l < (unsigned)NUM_CLASSES) ? l : 0;
    }
    __syncthreads();
    if (t < NUM_CLASSES) {
        float s = 0.f;
        for (int j = 0; j < KNN_K; ++j)
            if (wl[j] == t) s += wv[j];
        cls[t] = s;
    }
    __syncthreads();
    if (t < NUM_CLASSES) {
        float s = cls[t];
        int rank = 0;
        for (int j = 0; j < NUM_CLASSES; ++j) {
            float sj = cls[j];
            rank += (sj > s) || (sj == s && j < t);   // stable argsort(-scores)
        }
        out[(size_t)b * NUM_CLASSES + rank] = (float)t;
    }
}

// ---------------------------------------------------------------------------
extern "C" void kernel_launch(void* const* d_in, const int* in_sizes, int n_in,
                              void* d_out, int out_size, void* d_ws, size_t ws_size,
                              hipStream_t stream) {
    const float* x   = (const float*)d_in[0];   // [BQ, DIM]
    const float* xt  = (const float*)d_in[1];   // [NTRAIN, DIM]
    const int*   ytr = (const int*)d_in[2];     // [NTRAIN]
    float* out = (float*)d_out;

    // Workspace layout (bytes):
    //   xb   bf16 x        :  4 MiB
    //   tb   bf16 x_train  : 64 MiB
    //   sim  fp32 chunk    : 32 MiB
    //   topv fp32          :  4 MiB
    //   topi int32         :  4 MiB     total = 108 MiB
    char* ws = (char*)d_ws;
    unsigned short* xb = (unsigned short*)ws;
    unsigned short* tb = (unsigned short*)(ws + (size_t)4 * 1024 * 1024);
    float* sim  = (float*)(ws + (size_t)(4 + 64) * 1024 * 1024);
    float* topv = (float*)(ws + (size_t)(4 + 64 + 32) * 1024 * 1024);
    int*   topi = (int*)  (ws + (size_t)(4 + 64 + 32 + 4) * 1024 * 1024);

    {
        int n = BQ * DIM;
        knn_convert_bf16<<<(n + 255) / 256, 256, 0, stream>>>(x, xb, n);
    }
    {
        int n = NTRAIN * DIM;
        knn_convert_bf16<<<(n + 255) / 256, 256, 0, stream>>>(xt, tb, n);
    }

    for (int c = 0; c < NCHUNK; ++c) {
        dim3 grid(CHUNK / 128, BQ / 128);
        knn_gemm_bf16<<<grid, 256, 0, stream>>>(xb, tb, sim, c * CHUNK);
        knn_merge_topk<<<BQ, 256, 0, stream>>>(sim, topv, topi, c * CHUNK, c == 0);
    }

    knn_finalize<<<BQ, 256, 0, stream>>>(topv, topi, ytr, out);
}
